// Convs_76029511073934
// MI455X (gfx1250) — matrix operand order repaired and checked by reading, compiled, not run on hardware
//
#include <hip/hip_runtime.h>
#include <hip/hip_bf16.h>
#include <stdint.h>

typedef uint32_t u32;
typedef unsigned short u16;
typedef __attribute__((ext_vector_type(4)))  u32   u32x4;
typedef __attribute__((ext_vector_type(8)))  u16   u16x8;
typedef __attribute__((ext_vector_type(16))) __bf16 v16bf;
typedef __attribute__((ext_vector_type(8)))  float v8f;

#define HH   28
#define WW   28
#define HW   784
#define BATCH 8
#define NTAP  9
#define PADH  30   // H + 2*PAD

struct Desc { int i0, i1, i2, i3; float g0, g1, g2, g3; };

__device__ __forceinline__ u16 f2bf(float f) {
  u32 u = __float_as_uint(f);
  u32 r = (u + 0x7FFFu + ((u >> 16) & 1u)) >> 16;   // round-to-nearest-even
  return (u16)r;
}

// Async global->LDS copy (CDNA5). lds_byte_off is the byte offset within the
// workgroup's LDS allocation (the VDST operand); gptr is the per-lane 64-bit
// global address. lds_base is an *escape* operand: it carries the shared-mem
// object's address into the asm so the "memory" clobber is understood to
// cover LDS writes (otherwise LLVM proves smem is untouched and deletes the
// consumer ds_load fragments).
__device__ __forceinline__ void async_copy_b128(u32 lds_byte_off, const void* gptr,
                                                void* lds_base) {
  asm volatile("global_load_async_to_lds_b128 %0, %1, off"
               :: "v"(lds_byte_off), "v"(gptr), "v"(lds_base) : "memory");
}

// ---------------------------------------------------------------------------
// Kernel A: offset + mask convs fused into per-(pixel,tap) sampling descriptor
// ---------------------------------------------------------------------------
__global__ void __launch_bounds__(256) conv_offset_mask(
    const float* __restrict__ x, const float* __restrict__ wp,
    const float* __restrict__ bp, const float* __restrict__ wm,
    const float* __restrict__ bm, Desc* __restrict__ desc, int C) {
  int tid = blockIdx.x * blockDim.x + threadIdx.x;
  int total = BATCH * HW * NTAP;
  if (tid >= total) return;
  int n   = tid % NTAP;
  int pix = (tid / NTAP) % HW;
  int b   = tid / (NTAP * HW);
  int i = pix / WW, j = pix % WW;

  float accx = bp[n], accy = bp[NTAP + n], accm = bm[n];
  const float* wpx = wp + (size_t)n * C * 9;
  const float* wpy = wp + (size_t)(NTAP + n) * C * 9;
  const float* wmm = wm + (size_t)n * C * 9;
  const float* xb  = x + (size_t)b * C * HW;

  for (int c = 0; c < C; ++c) {
    const float* xc = xb + (size_t)c * HW;
    float win[9];
#pragma unroll
    for (int t = 0; t < 9; ++t) {
      int ii = i + t / 3 - 1, jj = j + t % 3 - 1;
      bool ok = (ii >= 0) & (ii < HH) & (jj >= 0) & (jj < WW);
      win[t] = ok ? xc[ii * WW + jj] : 0.f;
    }
    const float* wa = wpx + c * 9;
    const float* wb = wpy + c * 9;
    const float* wc = wmm + c * 9;
#pragma unroll
    for (int t = 0; t < 9; ++t) {
      accx = fmaf(win[t], wa[t], accx);
      accy = fmaf(win[t], wb[t], accy);
      accm = fmaf(win[t], wc[t], accm);
    }
  }
  float mval = 1.f / (1.f + __expf(-accm));
  float px = accx + (float)(i + 1) + (float)(n / 3 - 1);
  float py = accy + (float)(j + 1) + (float)(n % 3 - 1);
  float fx = floorf(px), fy = floorf(py);
  const float hi = (float)(PADH - 1);
  float ltx = fminf(fmaxf(fx, 0.f), hi),       lty = fminf(fmaxf(fy, 0.f), hi);
  float rbx = fminf(fmaxf(fx + 1.f, 0.f), hi), rby = fminf(fmaxf(fy + 1.f, 0.f), hi);
  float pxc = fminf(fmaxf(px, 0.f), hi),       pyc = fminf(fmaxf(py, 0.f), hi);
  float glt = (1.f + (ltx - pxc)) * (1.f + (lty - pyc)) * mval;
  float grb = (1.f - (rbx - pxc)) * (1.f - (rby - pyc)) * mval;
  float glb = (1.f + (ltx - pxc)) * (1.f - (rby - pyc)) * mval;
  float grt = (1.f - (rbx - pxc)) * (1.f + (lty - pyc)) * mval;

  Desc d;
  auto enc = [&](float qx, float qy, float& g) -> int {
    int xi = (int)qx, yi = (int)qy;   // padded coords; zero-pad border folds to g=0
    if (xi >= 1 && xi <= HH && yi >= 1 && yi <= WW) return (xi - 1) * WW + (yi - 1);
    g = 0.f; return 0;
  };
  d.g0 = glt; d.i0 = enc(ltx, lty, d.g0);
  d.g1 = grb; d.i1 = enc(rbx, rby, d.g1);
  d.g2 = glb; d.i2 = enc(ltx, rby, d.g2);
  d.g3 = grt; d.i3 = enc(rbx, lty, d.g3);
  desc[tid] = d;
}

// ---------------------------------------------------------------------------
// Kernel B: deformable im2col -> bf16 A matrix [Mrows][C*9]
// ---------------------------------------------------------------------------
__global__ void __launch_bounds__(256) deform_im2col(
    const float* __restrict__ xin, const Desc* __restrict__ desc,
    u16* __restrict__ A, int C, int m0, int Mrows) {
  int tid = blockIdx.x * blockDim.x + threadIdx.x;
  if (tid >= Mrows * C) return;
  int c = tid % C;
  int mloc = tid / C;
  int m = m0 + mloc;
  int b = m / HW;
  const Desc* d = desc + (size_t)m * NTAP;
  const float* xc = xin + ((size_t)b * C + c) * HW;
  size_t K = (size_t)C * NTAP;
  u16* out = A + (size_t)mloc * K + (size_t)c * NTAP;
#pragma unroll
  for (int n = 0; n < NTAP; ++n) {
    Desc dn = d[n];
    float v = dn.g0 * xc[dn.i0] + dn.g1 * xc[dn.i1] +
              dn.g2 * xc[dn.i2] + dn.g3 * xc[dn.i3];
    out[n] = f2bf(v);
  }
}

// ---------------------------------------------------------------------------
// Kernel C: weight [O][K] fp32 -> transposed bf16 [K][O]
// ---------------------------------------------------------------------------
__global__ void __launch_bounds__(256) weight_to_bf16_t(
    const float* __restrict__ w, u16* __restrict__ wt, int K, int NO) {
  int tid = blockIdx.x * blockDim.x + threadIdx.x;
  if (tid >= K * NO) return;
  int o = tid % NO, k = tid / NO;
  wt[(size_t)k * NO + o] = f2bf(w[(size_t)o * K + k]);
}

// ---------------------------------------------------------------------------
// Kernel D: bf16 WMMA GEMM. 256 thr = 8 wave32; macro tile 128x64;
// wave = 32x32 C block = 2x2 v_wmma_f32_16x16x32_bf16 tiles per K-step.
// Staging: double-buffered LDS fed by global_load_async_to_lds_b128
// (ASYNCcnt), overlapped with the WMMA compute phase. Last iteration peeled
// so the hot loop is branch-free.
// ---------------------------------------------------------------------------
union FragAB { u32x4 q[2]; v16bf v; };

#define A_ELEMS 4096          // 128*32 u16 per buffer
#define B_ELEMS 2048          // 32*64  u16 per buffer

__global__ void __launch_bounds__(256) wmma_gemm_bf16(
    const u16* __restrict__ A, const u16* __restrict__ Wt,
    float* __restrict__ out, int m0_global, int K, int NO, int relu) {
  // Single shared object => allocation starts at LDS offset 0, so byte
  // offsets used by the async-copy VDST operand match C-level indexing.
  __shared__ __attribute__((aligned(16))) u16 smem[2 * A_ELEMS + 2 * B_ELEMS];
  u16* AsmBuf = smem;                 // [2][A_ELEMS]
  u16* BsmBuf = smem + 2 * A_ELEMS;   // [2][B_ELEMS]

  int t    = threadIdx.x;
  int lane = t & 31;
  int wav  = t >> 5;
  int wm   = wav & 3;      // M wave offset: wm*32
  int wn   = wav >> 2;     // N wave offset: wn*32
  int lgrp = lane >> 4;    // lane half
  int lmod = lane & 15;

  size_t Arow0 = (size_t)blockIdx.x * 128;
  int n0 = blockIdx.y * 64;

  // global->LDS staging assignments (3 x 16B per thread per K-step)
  int arow  = t >> 1;            // 0..127
  int ahalf = (t & 1) * 16;      // 0 / 16 (k within 32-chunk)
  int brow  = t >> 3;            // 0..31  (k within 32-chunk)
  int bcol  = (t & 7) * 8;       // 0..56

  const u16* Aptr = A  + (Arow0 + arow) * (size_t)K + ahalf;
  const u16* Bptr = Wt + (size_t)brow * NO + n0 + bcol;

  u32 aoff0[2], aoff1[2], boff[2];   // LDS byte offsets per buffer
#pragma unroll
  for (int bb = 0; bb < 2; ++bb) {
    aoff0[bb] = (u32)((bb * A_ELEMS + arow * 32 + ahalf) * 2);
    aoff1[bb] = aoff0[bb] + 16;
    boff[bb]  = (u32)((2 * A_ELEMS + bb * B_ELEMS + brow * 64 + bcol) * 2);
  }

  v8f zero = {0.f, 0.f, 0.f, 0.f, 0.f, 0.f, 0.f, 0.f};
  v8f acc[2][2];
#pragma unroll
  for (int mi = 0; mi < 2; ++mi)
#pragma unroll
    for (int ni = 0; ni < 2; ++ni) acc[mi][ni] = zero;

  int KT = K >> 5;   // always >= 2 here (72 or 288)

  auto compute_tile = [&](int cur) {
    const u16* Acur = AsmBuf + cur * A_ELEMS;
    const u16* Bcur = BsmBuf + cur * B_ELEMS;
    FragAB af[2], bfr[2];
#pragma unroll
    for (int mi = 0; mi < 2; ++mi) {
      int r = (wm * 32 + mi * 16 + lmod) * 32;     // A row in LDS
      // VGPR0-3: K = lgrp*8 .. +7 ; VGPR4-7: K = 16+lgrp*8 .. +7
      af[mi].q[0] = *(const u32x4*)&Acur[r + lgrp * 8];
      af[mi].q[1] = *(const u32x4*)&Acur[r + 16 + lgrp * 8];
    }
#pragma unroll
    for (int ni = 0; ni < 2; ++ni) {
      int cb = wn * 32 + ni * 16;                  // B: N = lmod, K in table order
#pragma unroll
      for (int s = 0; s < 16; ++s) {
        int kk = (s < 8) ? (lgrp * 8 + s) : (16 + lgrp * 8 + (s - 8));
        ((u16*)&bfr[ni])[s] = Bcur[kk * 64 + cb + lmod];
      }
    }
#pragma unroll
    for (int mi = 0; mi < 2; ++mi)
#pragma unroll
      for (int ni = 0; ni < 2; ++ni)
        acc[mi][ni] = __builtin_amdgcn_wmma_f32_16x16x32_bf16(
            false, af[mi].v, false, bfr[ni].v, (short)0, acc[mi][ni],
            false, false);
  };

  // prologue: tile 0 into buffer 0
  async_copy_b128(aoff0[0], Aptr,     smem);
  async_copy_b128(aoff1[0], Aptr + 8, smem);
  async_copy_b128(boff[0],  Bptr,     smem);

  // steady state: issue kt+1, wait for kt, compute kt
  for (int kt = 0; kt < KT - 1; ++kt) {
    int cur = kt & 1, nxt = cur ^ 1;
    __syncthreads();                   // all waves done reading buf[nxt] (kt-1)
    const u16* An = Aptr + (size_t)(kt + 1) * 32;
    const u16* Bn = Bptr + (size_t)(kt + 1) * 32 * NO;
    __builtin_prefetch(An + 64, 0, 0);             // warm L2 two tiles ahead
    async_copy_b128(aoff0[nxt], An,     smem);
    async_copy_b128(aoff1[nxt], An + 8, smem);
    async_copy_b128(boff[nxt],  Bn,     smem);
    // 3 newest copies outstanding -> previous tile's 3 (in-order) are done
    asm volatile("s_wait_asynccnt 0x3" ::: "memory");
    __syncthreads();                   // buf[cur] complete across all waves
    compute_tile(cur);
  }
  // epilogue: last tile
  asm volatile("s_wait_asynccnt 0x0" ::: "memory");
  __syncthreads();
  compute_tile((KT - 1) & 1);

  // C layout: lane 0-15 -> M=VGPRr, lane 16-31 -> M=8+r; N = lane%16. Scatter NCHW.
#pragma unroll
  for (int mi = 0; mi < 2; ++mi) {
#pragma unroll
    for (int ni = 0; ni < 2; ++ni) {
      int col = n0 + wn * 32 + ni * 16 + lmod;
#pragma unroll
      for (int r = 0; r < 8; ++r) {
        int mg = m0_global + (int)Arow0 + wm * 32 + mi * 16 + lgrp * 8 + r;
        int bidx = mg / HW, pixq = mg % HW;
        float v = acc[mi][ni][r];
        if (relu) v = fmaxf(v, 0.f);
        out[((size_t)bidx * NO + col) * HW + pixq] = v;
      }
    }
  }
}

// ---------------------------------------------------------------------------
extern "C" void kernel_launch(void* const* d_in, const int* in_sizes, int n_in,
                              void* d_out, int out_size, void* d_ws, size_t ws_size,
                              hipStream_t stream) {
  const float* x    = (const float*)d_in[0];
  const float* w_p1 = (const float*)d_in[1];
  const float* b_p1 = (const float*)d_in[2];
  const float* w_m1 = (const float*)d_in[3];
  const float* b_m1 = (const float*)d_in[4];
  const float* w_c1 = (const float*)d_in[5];
  const float* w_p2 = (const float*)d_in[6];
  const float* b_p2 = (const float*)d_in[7];
  const float* w_m2 = (const float*)d_in[8];
  const float* b_m2 = (const float*)d_in[9];
  const float* w_c2 = (const float*)d_in[10];
  float* out = (float*)d_out;

  const int C1 = 256, C2 = 1024;
  const int K1 = C1 * NTAP;      // 2304
  const int K2 = C2 * NTAP;      // 9216
  const int M  = BATCH * HW;     // 6272 = 49*128

  char* ws = (char*)d_ws;
  size_t off = 0;
  auto alloc = [&](size_t bytes) {
    void* p = ws + off; off += (bytes + 255) & ~(size_t)255; return p;
  };
  Desc*  desc = (Desc*)alloc((size_t)M * NTAP * sizeof(Desc));  // 1.8 MB (shared L1/L2)
  u16*   Wt1  = (u16*) alloc((size_t)K1 * C2 * 2);              // 4.7 MB
  u16*   Wt2  = (u16*) alloc((size_t)K2 * C1 * 2);              // 4.7 MB
  float* hbuf = (float*)alloc((size_t)BATCH * C2 * HW * 4);     // 25.7 MB
  u16*   Abuf = (u16*) alloc((size_t)M * K1 * 2);               // 28.9 MB (reused for L2 chunks)
  (void)ws_size;

  // ---- layer 1 ----
  {
    int n = K1 * C2;
    weight_to_bf16_t<<<(n + 255) / 256, 256, 0, stream>>>(w_c1, Wt1, K1, C2);
  }
  {
    int n = M * NTAP;
    conv_offset_mask<<<(n + 255) / 256, 256, 0, stream>>>(x, w_p1, b_p1, w_m1, b_m1, desc, C1);
  }
  {
    int n = M * C1;
    deform_im2col<<<(n + 255) / 256, 256, 0, stream>>>(x, desc, Abuf, C1, 0, M);
  }
  {
    dim3 g(M / 128, C2 / 64);
    wmma_gemm_bf16<<<g, 256, 0, stream>>>(Abuf, Wt1, hbuf, 0, K1, C2, 1);
  }

  // ---- layer 2 ----
  {
    int n = K2 * C1;
    weight_to_bf16_t<<<(n + 255) / 256, 256, 0, stream>>>(w_c2, Wt2, K2, C1);
    int n2 = M * NTAP;
    conv_offset_mask<<<(n2 + 255) / 256, 256, 0, stream>>>(hbuf, w_p2, b_p2, w_m2, b_m2, desc, C2);
  }
  const int MCHUNK = 896;                    // 7*128 ; 6272 = 7 chunks
  for (int m0 = 0; m0 < M; m0 += MCHUNK) {
    int n = MCHUNK * C2;
    deform_im2col<<<(n + 255) / 256, 256, 0, stream>>>(hbuf, desc, Abuf, C2, m0, MCHUNK);
    dim3 g(MCHUNK / 128, C1 / 64);
    wmma_gemm_bf16<<<g, 256, 0, stream>>>(Abuf, Wt2, out, m0, K2, C1, 0);
  }
}
